// AdaFeatBlock_26336739459172
// MI455X (gfx1250) — matrix operand order repaired and checked by reading, not compiled, not run on hardware
//
#include <hip/hip_runtime.h>
#include <hip/hip_bf16.h>
#include <math.h>

// ---------------------------------------------------------------------------
// DCNv2 on MI455X (gfx1250): bf16 WMMA implicit-GEMM pipeline with
// async global->LDS staging (ASYNCcnt) and double-buffered tap tiles.
// ---------------------------------------------------------------------------

typedef __bf16 bf16;
typedef bf16  v16bf __attribute__((ext_vector_type(16)));
typedef bf16  v8bf  __attribute__((ext_vector_type(8)));
typedef float v8f   __attribute__((ext_vector_type(8)));
typedef float v4f   __attribute__((ext_vector_type(4)));

#define BB   8
#define CC   64
#define HH   128
#define WW   128
#define HWT  (HH * WW)
#define KFT  9
#define OMC  32   // 27 offset/mask channels padded to 32
#define OCH  64   // deform-conv output channels
#define ASTRIDE 72   // LDS A-tile row stride in halfs (144B = 9*16B, bank-friendly)
#define OSTRIDE 132  // LDS out-tile row stride in floats (528B = 33*16B)

// ---- CDNA5 async global<->LDS path (guarded: falls back to sync copies) ----
#if defined(__AMDGCN__) && __has_builtin(__builtin_amdgcn_global_load_async_to_lds_b128) && \
    __has_builtin(__builtin_amdgcn_s_wait_asynccnt)
#define USE_ASYNC_LD 1
#else
#define USE_ASYNC_LD 0
#endif
#if defined(__AMDGCN__) && __has_builtin(__builtin_amdgcn_global_store_async_from_lds_b128) && \
    __has_builtin(__builtin_amdgcn_s_wait_asynccnt)
#define USE_ASYNC_ST 1
#else
#define USE_ASYNC_ST 0
#endif

#if USE_ASYNC_LD || USE_ASYNC_ST
// The builtins take pointers to GCC-vector int4: AS1 (global) / AS3 (LDS).
// On AMDGPU the low 32 bits of a flat LDS address are the LDS offset
// (ISA 10.2), so the int-routed casts are exact.
typedef int v4i_vs __attribute__((vector_size(16)));
typedef __attribute__((address_space(1))) v4i_vs* gbl_v4i_p;
typedef __attribute__((address_space(3))) v4i_vs* lds_v4i_p;
#endif

#if USE_ASYNC_LD
template <int OFF>
__device__ __forceinline__ void async_ld_b128(const void* g, void* l) {
    __builtin_amdgcn_global_load_async_to_lds_b128(
        (gbl_v4i_p)(unsigned long long)g,
        (lds_v4i_p)(unsigned int)(unsigned long long)l,
        OFF, 0);
}
#endif
#if USE_ASYNC_ST
__device__ __forceinline__ void async_st_b128(void* g, const void* l) {
    __builtin_amdgcn_global_store_async_from_lds_b128(
        (gbl_v4i_p)(unsigned long long)g,
        (lds_v4i_p)(unsigned int)(unsigned long long)l,
        0, 0);
}
#endif

// ---------------------------------------------------------------------------
// Kernel 0: x NCHW fp32 -> NHWC bf16 (LDS tile transpose, coalesced R+W)
// ---------------------------------------------------------------------------
__global__ void xpose_kernel(const float* __restrict__ x, bf16* __restrict__ xb) {
    __shared__ float t[32][33];
    const int b  = blockIdx.z;
    const int p0 = blockIdx.x * 32;   // flattened spatial
    const int c0 = blockIdx.y * 32;   // channel
    const int tx = threadIdx.x;       // 0..31
    const int ty0 = threadIdx.y;      // 0..7
    #pragma unroll
    for (int ty = ty0; ty < 32; ty += 8)
        t[ty][tx] = x[(size_t)(b * CC + c0 + ty) * HWT + p0 + tx];
    __syncthreads();
    #pragma unroll
    for (int ty = ty0; ty < 32; ty += 8)
        xb[(size_t)(b * HWT + p0 + ty) * CC + c0 + tx] = (bf16)t[tx][ty];
}

// ---------------------------------------------------------------------------
// Kernel 1: weights -> [tap][out_ch][in_ch] bf16 (B-operand friendly layout)
// ---------------------------------------------------------------------------
__global__ void wprep_kernel(const float* __restrict__ w_om,
                             const float* __restrict__ w_dc,
                             bf16* __restrict__ w_om_b,
                             bf16* __restrict__ w_dc_b) {
    const int i = blockIdx.x * 256 + threadIdx.x;
    const int n_om = KFT * OMC * CC;
    if (i < n_om) {
        const int tap = i / (OMC * CC);
        const int r   = i % (OMC * CC);
        const int oc  = r / CC;
        const int c   = r % CC;
        float v = (oc < 27) ? w_om[(size_t)(oc * CC + c) * KFT + tap] : 0.f;
        w_om_b[i] = (bf16)v;
    } else {
        const int j = i - n_om;
        if (j < KFT * OCH * CC) {
            const int tap = j / (OCH * CC);
            const int r   = j % (OCH * CC);
            const int o   = r / CC;
            const int c   = r % CC;
            w_dc_b[j] = (bf16)w_dc[(size_t)(o * CC + c) * KFT + tap];
        }
    }
}

// Build a WMMA A-fragment from an LDS row (contiguous K order, matches B):
// lanes 0-15: K = k0+0..15 ; lanes 16-31: K = k0+16..31
__device__ __forceinline__ v16bf load_a_frag(const bf16* arow, int k0, int kb) {
    v8bf alo = *(const v8bf*)(arow + k0 + 2 * kb);
    v8bf ahi = *(const v8bf*)(arow + k0 + 2 * kb + 8);
    v16bf a;
    #pragma unroll
    for (int j = 0; j < 8; ++j) { a[j] = alo[j]; a[8 + j] = ahi[j]; }
    return a;
}

// Stage one (pixel, 32-channel half) slice of the shifted row tile for `tap`.
__device__ __forceinline__ void stage_tap(const bf16* __restrict__ xb, bf16* buf,
                                          int b, int y, int px, int ch0, int tap) {
    const int dy = tap / 3 - 1, dx = tap % 3 - 1;
    const int sy = y + dy;
    const int sx = px + dx;
    bf16* dst = buf + px * ASTRIDE + ch0;
    if (sy >= 0 && sy < HH && sx >= 0 && sx < WW) {
        const bf16* src = &xb[(size_t)((b * HH + sy) * WW + sx) * CC + ch0];
#if USE_ASYNC_LD
        async_ld_b128<0>(src, dst);
        async_ld_b128<16>(src, dst);
        async_ld_b128<32>(src, dst);
        async_ld_b128<48>(src, dst);
#else
        #pragma unroll
        for (int j = 0; j < 4; ++j)
            *(v8bf*)(dst + j * 8) = *(const v8bf*)(src + j * 8);
#endif
    } else {
        v8bf z = {};
        #pragma unroll
        for (int j = 0; j < 4; ++j)
            *(v8bf*)(dst + j * 8) = z;
    }
}

// ---------------------------------------------------------------------------
// Kernel 2: offset/mask conv (27ch, padded 32) as implicit GEMM.
// One workgroup per image row: 128 px, 8 waves x 16-px M-tiles, K = 9*64.
// Double-buffered async staging: tap t+1's DMA overlaps tap t's WMMAs.
// Output layout: om[(b*HW + p)*32 + ch]  (channel-contiguous -> coalesced)
// ---------------------------------------------------------------------------
__global__ __launch_bounds__(256)
void conv_om_kernel(const bf16* __restrict__ xb, const bf16* __restrict__ wb,
                    const float* __restrict__ bias, float* __restrict__ om) {
    __shared__ __attribute__((aligned(16))) bf16 Asub[2][128 * ASTRIDE];
    const int row  = blockIdx.x;          // 0 .. B*H-1
    const int b    = row / HH;
    const int y    = row % HH;
    const int tid  = threadIdx.x;
    const int lane = tid & 31;
    const int wave = tid >> 5;            // 0..7
    const int px   = tid >> 1;            // staging pixel
    const int ch0  = (tid & 1) * 32;      // staging channel half
    v8f acc0 = {}, acc1 = {};

    stage_tap(xb, Asub[0], b, y, px, ch0, 0);   // prologue: tap 0

    for (int tap = 0; tap < KFT; ++tap) {
#if USE_ASYNC_LD
        __builtin_amdgcn_s_wait_asynccnt(0);
#endif
        __syncthreads();   // Asub[tap&1] ready; all waves done with prior WMMAs
        if (tap + 1 < KFT)
            stage_tap(xb, Asub[(tap + 1) & 1], b, y, px, ch0, tap + 1);

        // 2 K-steps x 2 N-tiles of v_wmma_f32_16x16x32_bf16 on Asub[tap&1]
        const int m  = lane & 15;
        const int kb = (lane >> 4) * 8;
        const bf16* arow  = &Asub[tap & 1][(wave * 16 + m) * ASTRIDE];
        const bf16* Bbase = &wb[tap * OMC * CC];
        #pragma unroll
        for (int ks = 0; ks < 2; ++ks) {
            const int k0   = ks * 32;
            const int kcol = k0 + (lane >> 4) * 16;
            v16bf a = load_a_frag(arow, k0, kb);
            v16bf b0 = *(const v16bf*)&Bbase[(lane & 15) * CC + kcol];
            acc0 = __builtin_amdgcn_wmma_f32_16x16x32_bf16(
                false, a, false, b0, (short)0, acc0, false, false);
            v16bf b1 = *(const v16bf*)&Bbase[(16 + (lane & 15)) * CC + kcol];
            acc1 = __builtin_amdgcn_wmma_f32_16x16x32_bf16(
                false, a, false, b1, (short)0, acc1, false, false);
        }
    }
    // epilogue: bias + channel-contiguous store
    const int n  = lane & 15;
    const int mh = (lane >> 4) * 8;
    const float bias0 = bias[n];
    const float bias1 = (n + 16 < 27) ? bias[n + 16] : 0.f;
    const int p0 = y * WW + wave * 16;
    #pragma unroll
    for (int r = 0; r < 8; ++r) {
        const int p = p0 + mh + r;
        float* dst = &om[(size_t)(b * HWT + p) * 32];
        dst[n]      = acc0[r] + bias0;
        dst[n + 16] = acc1[r] + bias1;
    }
}

// ---------------------------------------------------------------------------
// Kernel 3: deformable conv. Per tap: bilinear-gather a 128x64 bf16 A-tile
// (L2-resident NHWC image, 1 cacheline per corner), then 8 WMMAs per wave.
// Epilogue transposes through LDS; output pushed with async LDS->global DMA.
// ---------------------------------------------------------------------------
__global__ __launch_bounds__(256)
void deform_kernel(const bf16* __restrict__ xb, const float* __restrict__ om,
                   const bf16* __restrict__ wb, const float* __restrict__ bias,
                   float* __restrict__ out) {
    __shared__ __attribute__((aligned(16))) bf16  Asub[128 * ASTRIDE];
    __shared__ __attribute__((aligned(16))) float Otile[OCH * OSTRIDE];
    const int row  = blockIdx.x;
    const int b    = row / HH;
    const int y    = row % HH;
    const int tid  = threadIdx.x;
    const int lane = tid & 31;
    const int wave = tid >> 5;
    v8f acc[4] = {};

    for (int tap = 0; tap < KFT; ++tap) {
        __syncthreads();
        {   // ---- gather phase: thread = (pixel, 32-channel half) ----
            const int px  = tid >> 1;
            const int ch0 = (tid & 1) * 32;
            const float* omp = &om[(size_t)(b * HWT + y * WW + px) * 32];
            const float oy = omp[2 * tap];
            const float ox = omp[2 * tap + 1];
            const float mask = 2.f / (1.f + __expf(-omp[18 + tap]));
            const float py  = (float)y  + (float)(tap / 3 - 1) + oy;
            const float pxf = (float)px + (float)(tap % 3 - 1) + ox;
            const float y0f = floorf(py), x0f = floorf(pxf);
            const float fy = py - y0f, fx = pxf - x0f;
            const int y0 = (int)y0f, x0 = (int)x0f;
            float accv[32];
            #pragma unroll
            for (int j = 0; j < 32; ++j) accv[j] = 0.f;
            #pragma unroll
            for (int cy = 0; cy < 2; ++cy) {
                const int   yy = y0 + cy;
                const float wy = cy ? fy : 1.f - fy;
                #pragma unroll
                for (int cx = 0; cx < 2; ++cx) {
                    const int   xx  = x0 + cx;
                    const float wgt = wy * (cx ? fx : 1.f - fx) * mask;
                    if (yy >= 0 && yy < HH && xx >= 0 && xx < WW && wgt != 0.f) {
                        const bf16* src =
                            &xb[(size_t)((b * HH + yy) * WW + xx) * CC + ch0];
                        #pragma unroll
                        for (int j4 = 0; j4 < 4; ++j4) {
                            v8bf v = *(const v8bf*)(src + j4 * 8);
                            #pragma unroll
                            for (int j = 0; j < 8; ++j)
                                accv[j4 * 8 + j] += wgt * (float)v[j];
                        }
                    }
                }
            }
            bf16* dst = &Asub[px * ASTRIDE + ch0];
            #pragma unroll
            for (int j = 0; j < 32; ++j) dst[j] = (bf16)accv[j];
        }
        __syncthreads();
        {   // ---- WMMA phase: 2 K-steps x 4 N-tiles ----
            const int m  = lane & 15;
            const int kb = (lane >> 4) * 8;
            const bf16* arow = &Asub[(wave * 16 + m) * ASTRIDE];
            const bf16* Bt   = &wb[tap * OCH * CC];
            #pragma unroll
            for (int ks = 0; ks < 2; ++ks) {
                const int k0   = ks * 32;
                const int kcol = k0 + (lane >> 4) * 16;
                v16bf a = load_a_frag(arow, k0, kb);
                #pragma unroll
                for (int nt = 0; nt < 4; ++nt) {
                    v16bf bb =
                        *(const v16bf*)&Bt[(nt * 16 + (lane & 15)) * CC + kcol];
                    acc[nt] = __builtin_amdgcn_wmma_f32_16x16x32_bf16(
                        false, a, false, bb, (short)0, acc[nt], false, false);
                }
            }
        }
    }
    __syncthreads();
    {   // accumulators -> LDS (transpose to channel-major), + bias
        const int n0 = lane & 15;
        const int mh = (lane >> 4) * 8;
        #pragma unroll
        for (int nt = 0; nt < 4; ++nt) {
            const int n = nt * 16 + n0;
            const float bn = bias[n];
            #pragma unroll
            for (int r = 0; r < 8; ++r)
                Otile[n * OSTRIDE + wave * 16 + mh + r] = acc[nt][r] + bn;
        }
    }
    __syncthreads();
    {   // NCHW store: chunk c = j*256+tid -> (n = c>>5, seg = c&31);
        // per wave-instruction: one channel row, 32 lanes x 16B = 512B contig.
        #pragma unroll
        for (int j = 0; j < 8; ++j) {
            const int c   = j * 256 + tid;
            const int n   = c >> 5;
            const int seg = c & 31;
            float* g = &out[(size_t)((b * OCH + n) * HH + y) * WW + seg * 4];
            const float* l = &Otile[n * OSTRIDE + seg * 4];
#if USE_ASYNC_ST
            async_st_b128(g, l);
#else
            *(v4f*)g = *(const v4f*)l;
#endif
        }
#if USE_ASYNC_ST
        __builtin_amdgcn_s_wait_asynccnt(0);
#endif
    }
}

// ---------------------------------------------------------------------------
extern "C" void kernel_launch(void* const* d_in, const int* in_sizes, int n_in,
                              void* d_out, int out_size, void* d_ws, size_t ws_size,
                              hipStream_t stream) {
    const float* x    = (const float*)d_in[0];
    const float* w_om = (const float*)d_in[1];
    const float* b_om = (const float*)d_in[2];
    const float* w_dc = (const float*)d_in[3];
    const float* b_dc = (const float*)d_in[4];
    float* out = (float*)d_out;

    char* ws = (char*)d_ws;
    size_t off = 0;
    bf16* xb = (bf16*)(ws + off);
    off += (size_t)BB * HWT * CC * sizeof(bf16);  off = (off + 255) & ~(size_t)255;
    float* om = (float*)(ws + off);
    off += (size_t)BB * HWT * 32 * sizeof(float); off = (off + 255) & ~(size_t)255;
    bf16* womb = (bf16*)(ws + off);
    off += (size_t)KFT * OMC * CC * sizeof(bf16); off = (off + 255) & ~(size_t)255;
    bf16* wdcb = (bf16*)(ws + off);

    xpose_kernel<<<dim3(HWT / 32, CC / 32, BB), dim3(32, 8), 0, stream>>>(x, xb);
    const int nprep = KFT * OMC * CC + KFT * OCH * CC;
    wprep_kernel<<<dim3((nprep + 255) / 256), dim3(256), 0, stream>>>(w_om, w_dc, womb, wdcb);
    conv_om_kernel<<<dim3(BB * HH), dim3(256), 0, stream>>>(xb, womb, b_om, om);
    deform_kernel<<<dim3(BB * HH), dim3(256), 0, stream>>>(xb, om, wdcb, b_dc, out);
}
